// FFReModel_64295660421392
// MI455X (gfx1250) — matrix operand order repaired and checked by reading, not compiled, not run on hardware
//
#include <hip/hip_runtime.h>
#include <hip/hip_bf16.h>

// ---------------- model dims ----------------
#define BB   4
#define LL   1024
#define DD   768
#define HH   12
#define DHH  64
#define FF   3072
#define VV   50257
#define MM   (BB*LL)      // 4096 tokens
#define NLAY 2
#define S_C  512
#define S_C2 256
#define S_Q  128
#define S_R  128

typedef __attribute__((ext_vector_type(16))) _Float16 v16h;
typedef __attribute__((ext_vector_type(8)))  float    v8f;

__device__ __forceinline__ v8f wmma_f16(v16h a, v16h b, v8f c) {
  return __builtin_amdgcn_wmma_f32_16x16x32_f16(
      /*neg_a=*/false, a, /*neg_b=*/false, b,
      /*c_mod=*/(short)0, c, /*reuse_a=*/false, /*reuse_b=*/false);
}

// ---------------- prep kernels ----------------
__global__ void k_convert_f16(const float* __restrict__ s, _Float16* __restrict__ d, int n) {
  int i = blockIdx.x * blockDim.x + threadIdx.x;
  if (i < n) d[i] = (_Float16)s[i];
}

// Pack weight into WMMA B-fragment order:
//   dst[(((kt*NT)+nt)*32 + lane)*16 + e] = W[k][n]
//   k = kt*32 + (lane<16 ? e : 16+e), n = nt*16 + (lane&15)
// transpose=0: src is [K,N] row-major. transpose=1: src is [N,K] row-major (W = src^T).
// n >= N padded with zeros (Npad = NT*16, multiple of 64).
__global__ void k_pack_b(const float* __restrict__ src, _Float16* __restrict__ dst,
                         int K, int N, int NT, int transpose) {
  int i = blockIdx.x * blockDim.x + threadIdx.x;
  int total = (K >> 5) * NT * 512;
  if (i >= total) return;
  int t2   = i >> 9;          // (kt*NT + nt)
  int r    = i & 511;
  int lane = r >> 4;
  int e    = r & 15;
  int kt   = t2 / NT;
  int nt   = t2 - kt * NT;
  int k    = kt * 32 + ((lane < 16) ? e : 16 + e);
  int n    = nt * 16 + (lane & 15);
  float v = 0.f;
  if (n < N) v = transpose ? src[(size_t)n * K + k] : src[(size_t)k * N + n];
  dst[i] = (_Float16)v;
}

// ---------------- ragged embed concat ----------------
__global__ void k_embed(const int* __restrict__ ctx, const int* __restrict__ c2,
                        const int* __restrict__ query, const int* __restrict__ resp,
                        const int* __restrict__ cl, const int* __restrict__ c2l,
                        const int* __restrict__ ql, const int* __restrict__ rl,
                        const float* __restrict__ wte, _Float16* __restrict__ out) {
  int b = blockIdx.x / LL;
  int p = blockIdx.x - b * LL;
  int s1 = cl[b], s2 = s1 + c2l[b], s3 = s2 + ql[b], s4 = s3 + rl[b];
  int tok = -1;
  if      (p < s1) tok = ctx[b * S_C + p];
  else if (p < s2) tok = c2[b * S_C2 + (p - s1)];
  else if (p < s3) tok = query[b * S_Q + (p - s2)];
  else if (p < s4) tok = resp[b * S_R + (p - s3)];
  _Float16* o = out + (size_t)blockIdx.x * DD;
  if (tok < 0) {
    for (int d = threadIdx.x; d < DD; d += blockDim.x) o[d] = (_Float16)0.f;
  } else {
    const float* e = wte + (size_t)tok * DD;
    for (int d = threadIdx.x; d < DD; d += blockDim.x) o[d] = (_Float16)e[d];
  }
}

// ---------------- generic WMMA GEMM (packed-B weights) ----------------
// C[M,N] = A[M,K] @ W[K,N] (+bias) (+res[(m%res_mod),N]) (+gelu); M%128==0, K%32==0
// Wp is B-fragment-packed with NT = Npad/16 tiles (Npad % 64 == 0).
// Block tile 128x64: 4 waves, each wave 32 rows x 64 cols (2 A frags reuse each B frag).
__global__ __launch_bounds__(128) void k_gemm(
    const _Float16* __restrict__ A, const _Float16* __restrict__ Wp,
    const float* __restrict__ bias, const float* __restrict__ res, int res_mod,
    float* __restrict__ outF32, _Float16* __restrict__ outF16,
    int M, int N, int NT, int K, int act) {
  int lane = threadIdx.x & 31;
  int wave = threadIdx.x >> 5;
  int m0 = blockIdx.y * 128 + wave * 32;
  int n0 = blockIdx.x * 64;
  int ntile0 = blockIdx.x * 4;
  v8f acc[2][4] = {};
  const _Float16* pa0 = A + (size_t)(m0 + (lane & 15)) * K + ((lane < 16) ? 0 : 16);
  const _Float16* pa1 = pa0 + (size_t)16 * K;
  const v16h*     pb  = (const v16h*)Wp + (size_t)ntile0 * 32 + lane;
  const size_t bstep = (size_t)NT * 32;
  for (int kt = 0; kt < (K >> 5); ++kt) {
    v16h a0, a1;
#pragma unroll
    for (int e = 0; e < 8; ++e) {
      a0[e] = pa0[e]; a0[e + 8] = pa0[e + 8];
      a1[e] = pa1[e]; a1[e + 8] = pa1[e + 8];
    }
#pragma unroll
    for (int nt = 0; nt < 4; ++nt) {
      v16h b = pb[nt * 32];
      acc[0][nt] = wmma_f16(a0, b, acc[0][nt]);
      acc[1][nt] = wmma_f16(a1, b, acc[1][nt]);
    }
    pa0 += 32;
    pa1 += 32;
    pb  += bstep;
  }
#pragma unroll
  for (int mi = 0; mi < 2; ++mi) {
#pragma unroll
    for (int nt = 0; nt < 4; ++nt) {
      int n = n0 + nt * 16 + (lane & 15);
      if (n >= N) continue;
#pragma unroll
      for (int r = 0; r < 8; ++r) {
        int m = m0 + mi * 16 + r + ((lane < 16) ? 0 : 8);
        float v = acc[mi][nt][r];
        if (bias) v += bias[n];
        if (res)  v += res[(size_t)(m % res_mod) * N + n];
        if (act == 1) {  // tanh GELU
          float x = v;
          v = 0.5f * x * (1.f + tanhf(0.7978845608f * (x + 0.044715f * x * x * x)));
        }
        if (outF32) outF32[(size_t)m * N + n] = v;
        if (outF16) outF16[(size_t)m * N + n] = (_Float16)v;
      }
    }
  }
}

// ---------------- layernorm (f32 in -> f16 out) ----------------
__global__ __launch_bounds__(256) void k_layernorm(const float* __restrict__ in,
                                                   const float* __restrict__ g,
                                                   const float* __restrict__ bta,
                                                   _Float16* __restrict__ out) {
  __shared__ float red0[256], red1[256];
  int m = blockIdx.x, tid = threadIdx.x;
  const float* row = in + (size_t)m * DD;
  float s = 0.f, ss = 0.f;
  for (int d = tid; d < DD; d += 256) { float x = row[d]; s += x; ss += x * x; }
  red0[tid] = s; red1[tid] = ss; __syncthreads();
  for (int st = 128; st > 0; st >>= 1) {
    if (tid < st) { red0[tid] += red0[tid + st]; red1[tid] += red1[tid + st]; }
    __syncthreads();
  }
  float mean = red0[0] * (1.f / DD);
  float var  = red1[0] * (1.f / DD) - mean * mean;
  float inv  = rsqrtf(var + 1e-5f);
  for (int d = tid; d < DD; d += 256)
    out[(size_t)m * DD + d] = (_Float16)((row[d] - mean) * inv * g[d] + bta[d]);
}

// ---------------- qkv repack: [M,3D] -> Q,K [B,H,T,DH], V^T [B,H,DH,T] ----------------
__global__ void k_repack_qkv(const _Float16* __restrict__ qkv, _Float16* __restrict__ q,
                             _Float16* __restrict__ k, _Float16* __restrict__ vT) {
  int i = blockIdx.x * blockDim.x + threadIdx.x;
  if (i >= MM * 3 * DD) return;
  int m = i / (3 * DD), col = i - m * (3 * DD);
  int which = col / DD, hc = (col % DD) / DHH, dh = col & (DHH - 1);
  int b = m / LL, t = m - b * LL;
  _Float16 val = qkv[i];
  if (which == 2) {
    vT[(((size_t)(b * HH + hc)) * DHH + dh) * LL + t] = val;          // transposed
  } else {
    size_t dst = (((size_t)(b * HH + hc)) * LL + t) * DHH + dh;
    if (which == 0) q[dst] = val; else k[dst] = val;
  }
}

// ---------------- fused attention: one wave per (b,h,16-row q tile) ----------------
__global__ __launch_bounds__(32) void k_attn(const _Float16* __restrict__ Q,
                                             const _Float16* __restrict__ Km,
                                             const _Float16* __restrict__ VT,
                                             const int* __restrict__ cl, const int* __restrict__ c2l,
                                             const int* __restrict__ ql, const int* __restrict__ rl,
                                             _Float16* __restrict__ O) {
  __shared__ float sc[16 * LL];            // 64 KB score slab
  _Float16* hp = (_Float16*)sc;            // in-place f16 probs (row stride 2048 halfs)
  int lane = threadIdx.x;
  int qt = blockIdx.x & 63;
  int h  = (blockIdx.x >> 6) % HH;
  int b  = blockIdx.x / (64 * HH);
  int total = cl[b] + c2l[b] + ql[b] + rl[b];
  const _Float16* qp = Q  + ((size_t)(b * HH + h)) * LL * DHH;
  const _Float16* kp = Km + ((size_t)(b * HH + h)) * LL * DHH;
  const _Float16* vp = VT + ((size_t)(b * HH + h)) * DHH * LL;   // [DH,T]

  // Q fragments (16 rows x 64 head dims -> two K=32 chunks)
  v16h a0, a1;
  {
    int m = qt * 16 + (lane & 15);
    int kb = (lane < 16) ? 0 : 16;
    const _Float16* p = qp + (size_t)m * DHH + kb;
#pragma unroll
    for (int e = 0; e < 8; ++e) {
      a0[e] = p[e];      a0[e + 8] = p[e + 8];
      a1[e] = p[32 + e]; a1[e + 8] = p[32 + e + 8];
    }
  }
  // scores = Q K^T * 1/sqrt(64), masked
  for (int kt = 0; kt < 64; ++kt) {
    v16h b0, b1;
    {
      int key = kt * 16 + (lane & 15);
      int db  = (lane < 16) ? 0 : 16;
      const _Float16* p = kp + (size_t)key * DHH + db;
#pragma unroll
      for (int e = 0; e < 16; ++e) { b0[e] = p[e]; b1[e] = p[32 + e]; }
    }
    v8f c = {};
    c = wmma_f16(a0, b0, c);
    c = wmma_f16(a1, b1, c);
    int col = kt * 16 + (lane & 15);
#pragma unroll
    for (int r = 0; r < 8; ++r) {
      int row  = r + ((lane < 16) ? 0 : 8);
      int qpos = qt * 16 + row;
      float sv = c[r] * 0.125f;
      if (col > qpos || col >= total) sv = -1e9f;
      sc[row * LL + col] = sv;
    }
  }
  __syncthreads();
  // wave-local softmax: 2 lanes per row for reductions, 1 lane per row for in-place f16 write
  {
    int row = lane >> 1, st = (lane & 1) * 512;
    float mx = -3.4e38f;
    for (int j = 0; j < 512; ++j) mx = fmaxf(mx, sc[row * LL + st + j]);
    mx = fmaxf(mx, __shfl_xor(mx, 1, 32));
    float sum = 0.f;
    for (int j = 0; j < 512; ++j) sum += expf(sc[row * LL + st + j] - mx);
    sum += __shfl_xor(sum, 1, 32);
    float mxr  = __shfl(mx, lane * 2, 32);
    float invr = __shfl(1.f / sum, lane * 2, 32);
    __syncthreads();
    if (lane < 16) {
      int r2 = lane;
      for (int j = 0; j < 1024; ++j) {
        float x = sc[r2 * LL + j];
        hp[r2 * 2048 + j] = (_Float16)(expf(x - mxr) * invr);  // safe: half j overlays float j/2
      }
    }
  }
  __syncthreads();
  // O = P @ V, 32-key chunks, 4 head-dim column tiles (V^T rows are contiguous in key)
  v8f o[4] = {};
  for (int ch = 0; ch < 32; ++ch) {
    v16h pa;
    {
      int m = lane & 15, kb = (lane < 16) ? 0 : 16;
      const _Float16* pp = hp + m * 2048 + ch * 32 + kb;
#pragma unroll
      for (int e = 0; e < 8; ++e) { pa[e] = pp[e]; pa[e + 8] = pp[e + 8]; }
    }
#pragma unroll
    for (int dt = 0; dt < 4; ++dt) {
      v16h vb;
      int dh   = dt * 16 + (lane & 15);
      int keyb = ch * 32 + ((lane < 16) ? 0 : 16);
      const _Float16* p = vp + (size_t)dh * LL + keyb;
#pragma unroll
      for (int e = 0; e < 16; ++e) vb[e] = p[e];
      o[dt] = wmma_f16(pa, vb, o[dt]);
    }
  }
  // store O into [B,T,H*DH] f16 (proj GEMM input layout)
  {
    int nc = lane & 15;
#pragma unroll
    for (int r = 0; r < 8; ++r) {
      int row = r + ((lane < 16) ? 0 : 8);
      size_t base = ((size_t)(b * LL + qt * 16 + row)) * DD + h * DHH;
#pragma unroll
      for (int dt = 0; dt < 4; ++dt)
        O[base + dt * 16 + nc] = (_Float16)o[dt][r];
    }
  }
}

// ---------------- host orchestration ----------------
extern "C" void kernel_launch(void* const* d_in, const int* in_sizes, int n_in,
                              void* d_out, int out_size, void* d_ws, size_t ws_size,
                              hipStream_t stream) {
  const int*   ctx    = (const int*)d_in[0];
  const int*   query  = (const int*)d_in[1];
  const int*   resp   = (const int*)d_in[2];
  const int*   c2     = (const int*)d_in[3];
  const int*   c_lens = (const int*)d_in[4];
  const int*   q_lens = (const int*)d_in[5];
  const int*   r_lens = (const int*)d_in[6];
  const int*   c2_lens= (const int*)d_in[7];
  const float* wte    = (const float*)d_in[8];
  const float* wpe    = (const float*)d_in[9];
  const float* ln1_g  = (const float*)d_in[10];
  const float* ln1_b  = (const float*)d_in[11];
  const float* wqkv   = (const float*)d_in[12];
  const float* bqkv   = (const float*)d_in[13];
  const float* wo     = (const float*)d_in[14];
  const float* bo     = (const float*)d_in[15];
  const float* ln2_g  = (const float*)d_in[16];
  const float* ln2_b  = (const float*)d_in[17];
  const float* wfc    = (const float*)d_in[18];
  const float* bfc    = (const float*)d_in[19];
  const float* wpr    = (const float*)d_in[20];
  const float* bpr    = (const float*)d_in[21];
  const float* lnf_g  = (const float*)d_in[22];
  const float* lnf_b  = (const float*)d_in[23];
  const float* ll_w   = (const float*)d_in[24];
  const float* ll_b   = (const float*)d_in[25];

  // padded N tile counts (Npad multiple of 64)
  const int NT_D  = DD / 16;                       // 48
  const int NT_3D = (3 * DD) / 16;                 // 144
  const int NT_F  = FF / 16;                       // 192
  const int NT_V  = ((VV + 63) / 64) * 4;          // 3144  (Npad = 50304)

  char* ws = (char*)d_ws; size_t off = 0;
  auto alloc = [&](size_t bytes) -> void* {
    void* p = ws + off; off = (off + bytes + 255) & ~(size_t)255; return p;
  };
  _Float16* wteT_p = (_Float16*)alloc((size_t)DD * NT_V * 16 * 2);           // packed [768, 50304]
  _Float16* llw_p  = (_Float16*)alloc((size_t)DD * DD * 2);
  _Float16* wqkv_p = (_Float16*)alloc((size_t)NLAY * DD * 3 * DD * 2);
  _Float16* wo_p   = (_Float16*)alloc((size_t)NLAY * DD * DD * 2);
  _Float16* wfc_p  = (_Float16*)alloc((size_t)NLAY * DD * FF * 2);
  _Float16* wpr_p  = (_Float16*)alloc((size_t)NLAY * FF * DD * 2);
  _Float16* cpri   = (_Float16*)alloc((size_t)MM * DD * 2);
  float*    hbuf   = (float*)   alloc((size_t)MM * DD * 4);
  _Float16* xh     = (_Float16*)alloc((size_t)MM * DD * 2);
  _Float16* qkv_h  = (_Float16*)alloc((size_t)MM * 3 * DD * 2);
  _Float16* qh     = (_Float16*)alloc((size_t)BB * HH * LL * DHH * 2);
  _Float16* kh     = (_Float16*)alloc((size_t)BB * HH * LL * DHH * 2);
  _Float16* vhT    = (_Float16*)alloc((size_t)BB * HH * DHH * LL * 2);
  _Float16* oh     = (_Float16*)alloc((size_t)MM * DD * 2);
  _Float16* gh     = (_Float16*)alloc((size_t)MM * FF * 2);

  auto cdiv = [](long long a, long long b) { return (int)((a + b - 1) / b); };

  // weight prep: pack into B-fragment order (f16)
  k_pack_b<<<cdiv((size_t)DD * NT_V * 16, 256), 256, 0, stream>>>(wte,  wteT_p, DD, VV, NT_V, 1);
  k_pack_b<<<cdiv((size_t)DD * DD, 256), 256, 0, stream>>>(ll_w, llw_p, DD, DD, NT_D, 1);
  for (int l = 0; l < NLAY; ++l) {
    k_pack_b<<<cdiv((size_t)DD * 3 * DD, 256), 256, 0, stream>>>(
        wqkv + (size_t)l * DD * 3 * DD, wqkv_p + (size_t)l * DD * 3 * DD, DD, 3 * DD, NT_3D, 0);
    k_pack_b<<<cdiv((size_t)DD * DD, 256), 256, 0, stream>>>(
        wo + (size_t)l * DD * DD, wo_p + (size_t)l * DD * DD, DD, DD, NT_D, 0);
    k_pack_b<<<cdiv((size_t)DD * FF, 256), 256, 0, stream>>>(
        wfc + (size_t)l * DD * FF, wfc_p + (size_t)l * DD * FF, DD, FF, NT_F, 0);
    k_pack_b<<<cdiv((size_t)FF * DD, 256), 256, 0, stream>>>(
        wpr + (size_t)l * FF * DD, wpr_p + (size_t)l * FF * DD, FF, DD, NT_D, 0);
  }

  // ragged embed concat (f16)
  k_embed<<<BB * LL, 256, 0, stream>>>(ctx, c2, query, resp, c_lens, c2_lens, q_lens, r_lens,
                                       wte, cpri);

  // h = cpri @ ll_w^T + ll_b + wpe[t]   (wpe folded as row-broadcast residual, res_mod = L)
  k_gemm<<<dim3(NT_D / 4, MM / 128), 128, 0, stream>>>(
      cpri, llw_p, ll_b, wpe, LL, hbuf, (_Float16*)nullptr, MM, DD, NT_D, DD, 0);

  for (int l = 0; l < NLAY; ++l) {
    k_layernorm<<<MM, 256, 0, stream>>>(hbuf, ln1_g + l * DD, ln1_b + l * DD, xh);
    // qkv = x @ wqkv + bqkv  -> f16
    k_gemm<<<dim3(NT_3D / 4, MM / 128), 128, 0, stream>>>(
        xh, wqkv_p + (size_t)l * DD * 3 * DD, bqkv + l * 3 * DD,
        (const float*)nullptr, 1, (float*)nullptr, qkv_h, MM, 3 * DD, NT_3D, DD, 0);
    k_repack_qkv<<<cdiv(MM * 3 * DD, 256), 256, 0, stream>>>(qkv_h, qh, kh, vhT);
    k_attn<<<BB * HH * (LL / 16), 32, 0, stream>>>(qh, kh, vhT, c_lens, c2_lens, q_lens, r_lens, oh);
    // h = h + o @ wo + bo   (in-place residual)
    k_gemm<<<dim3(NT_D / 4, MM / 128), 128, 0, stream>>>(
        oh, wo_p + (size_t)l * DD * DD, bo + l * DD, hbuf, MM, hbuf, (_Float16*)nullptr,
        MM, DD, NT_D, DD, 0);
    k_layernorm<<<MM, 256, 0, stream>>>(hbuf, ln2_g + l * DD, ln2_b + l * DD, xh);
    // g = gelu(x @ wfc + bfc) -> f16
    k_gemm<<<dim3(NT_F / 4, MM / 128), 128, 0, stream>>>(
        xh, wfc_p + (size_t)l * DD * FF, bfc + l * FF,
        (const float*)nullptr, 1, (float*)nullptr, gh, MM, FF, NT_F, DD, 1);
    // h = h + g @ wpr + bpr   (in-place residual)
    k_gemm<<<dim3(NT_D / 4, MM / 128), 128, 0, stream>>>(
        gh, wpr_p + (size_t)l * FF * DD, bpr + l * DD, hbuf, MM, hbuf, (_Float16*)nullptr,
        MM, DD, NT_D, FF, 0);
  }

  // final LN then tied LM head: logits = x @ wte^T  -> d_out (f32)
  k_layernorm<<<MM, 256, 0, stream>>>(hbuf, lnf_g, lnf_b, xh);
  k_gemm<<<dim3(NT_V / 4, MM / 128), 128, 0, stream>>>(
      xh, wteT_p, (const float*)nullptr, (const float*)nullptr, 1,
      (float*)d_out, (_Float16*)nullptr, MM, VV, NT_V, DD, 0);
}
